// LigandEncoder_1065151889766
// MI455X (gfx1250) — matrix operand order repaired and checked by reading, not compiled, not run on hardware
//
#include <hip/hip_runtime.h>
#include <hip/hip_bf16.h>

// ---------------------------------------------------------------------------
// GINE encoder for MI455X (gfx1250, wave32, WMMA, async-LDS staging).
//
// Per layer:
//   zero(agg, stats)
//   edge_msg:    agg[dst] += relu(h[src] + edge_attr@We + be)   (native f32 atomics)
//   add_convert: abuf = bf16(h + agg)                            (A operand)
//   mlp_fused:   A tile -> LDS via GLOBAL_LOAD_ASYNC_TO_LDS_B128,
//                z2 = relu(A@W1+b1)@W2+b2 -> h (v_wmma_f32_16x16x32_bf16),
//                BN partial sums reduced in LDS, flushed with global atomics
//   bn_apply:    h = relu(gamma*(z2-mu)*rsqrt(var+eps)+beta)
// Final: global mean pool via atomics.
// ---------------------------------------------------------------------------

#define N_NODES  200000   // == 64 * 3125 (no GEMM boundary guards needed)
#define N_EDGES  400000   // == 8 * 50000
#define N_GRAPHS 8192
#define N_LAYERS 4
#define HID      256
#define BN_EPS   1e-5f

typedef __attribute__((ext_vector_type(16))) __bf16          v16bf;
typedef __attribute__((ext_vector_type(8)))  float           v8f;
typedef __attribute__((ext_vector_type(8)))  unsigned short  u16x8;
typedef __attribute__((ext_vector_type(4)))  int             b128_t;

union BfPack { v16bf v; u16x8 h[2]; unsigned short s[16]; };

__device__ __forceinline__ unsigned short f2bf(float f) {
  unsigned u = __builtin_bit_cast(unsigned, f);
  u += 0x7FFFu + ((u >> 16) & 1u);          // round-to-nearest-even
  return (unsigned short)(u >> 16);
}

__device__ __forceinline__ void gatomic_add(float* p, float v) {
  __hip_atomic_fetch_add(p, v, __ATOMIC_RELAXED, __HIP_MEMORY_SCOPE_AGENT);
}

// --------------------------------- utility ---------------------------------

__global__ void zero_f32(float* __restrict__ p, long n) {
  long i = (long)blockIdx.x * blockDim.x + threadIdx.x;
  long stride = (long)gridDim.x * blockDim.x;
  for (; i < n; i += stride) p[i] = 0.0f;
}

// Repack w1/w2 (f32 [L][K=256][N=256] row-major) into the WMMA B-operand
// per-lane layout: pack[l][(kt*256 + n)*32 + (k%32)] so a lane (col n, K-half)
// reads its 16 bf16 K-values as two contiguous b128 loads.
__global__ void repack_w(const float* __restrict__ w1, const float* __restrict__ w2,
                         unsigned short* __restrict__ p1, unsigned short* __restrict__ p2) {
  long tid = (long)blockIdx.x * blockDim.x + threadIdx.x;   // over L*256*256
  long l   = tid >> 16;
  int  k   = (int)((tid >> 8) & 255);
  int  n   = (int)(tid & 255);
  long dst = (l << 16) + (long)(((k >> 5) * HID + n) * 32 + (k & 31));
  p1[dst] = f2bf(w1[tid]);
  p2[dst] = f2bf(w2[tid]);
}

// ------------------------------ input projection ----------------------------
__global__ __launch_bounds__(256) void input_proj(
    const float* __restrict__ x, const float* __restrict__ W,
    const float* __restrict__ b, float* __restrict__ h) {
  __shared__ float sx[8][14];
  int wave = threadIdx.x >> 5, lane = threadIdx.x & 31;
  long node = (long)blockIdx.x * 8 + wave;
  if (lane < 14) sx[wave][lane] = x[node * 14 + lane];
  __syncthreads();
#pragma unroll
  for (int i = 0; i < 8; i++) {
    int c = i * 32 + lane;
    float acc = b[c];
#pragma unroll
    for (int k = 0; k < 14; k++) acc = fmaf(sx[wave][k], W[k * HID + c], acc);
    h[node * HID + c] = acc;
  }
}

// ------------------------------ edge messages -------------------------------
__global__ __launch_bounds__(256) void edge_msg(
    const float* __restrict__ ea, const long long* __restrict__ ei,
    const float* __restrict__ eW, const float* __restrict__ eb,
    const float* __restrict__ h, float* __restrict__ agg) {
  __shared__ float sW0[HID], sW1[HID], sW2[HID], sb[HID];
  int t = threadIdx.x;
  sW0[t] = eW[t]; sW1[t] = eW[HID + t]; sW2[t] = eW[2 * HID + t]; sb[t] = eb[t];
  __syncthreads();
  int wave = t >> 5, lane = t & 31;
  long e = (long)blockIdx.x * 8 + wave;
  int s = (int)ei[e];
  int d = (int)ei[N_EDGES + e];
  float a0 = ea[e * 3 + 0], a1 = ea[e * 3 + 1], a2 = ea[e * 3 + 2];
  const float* hs = h + (long)s * HID;
  float* ad = agg + (long)d * HID;
#pragma unroll
  for (int i = 0; i < 8; i++) {
    int c = i * 32 + lane;
    float m = hs[c] + a0 * sW0[c] + a1 * sW1[c] + a2 * sW2[c] + sb[c];
    gatomic_add(&ad[c], fmaxf(m, 0.0f));
  }
}

// ------------------------- z = h + agg, convert to bf16 ---------------------
__global__ void add_convert(const float2* __restrict__ h, const float2* __restrict__ agg,
                            unsigned int* __restrict__ abuf, long n2) {
  long i = (long)blockIdx.x * blockDim.x + threadIdx.x;
  long stride = (long)gridDim.x * blockDim.x;
  for (; i < n2; i += stride) {
    float2 a = h[i], b = agg[i];
    unsigned lo = f2bf(a.x + b.x), hi = f2bf(a.y + b.y);
    abuf[i] = lo | (hi << 16);
  }
}

// ------------------------------- fused MLP ----------------------------------
// 64-row stripe per workgroup (256 thr = 8 waves). Wave w: row-tile rt=w/2,
// col-tiles (w&1)*8 .. +7.
// Phase 0: async-stage A tile (64x256 bf16 = 32KB) into LDS.
// Phase 1: GEMM1 -> relu -> bf16 t tile in LDS.
// Phase 2: GEMM2 from t tile -> z2 global + BN partial sums.
// LDS: alds 32KB + tlds 32KB = 64KB; BN scratch aliases alds after GEMM1.
__global__ __launch_bounds__(256) void mlp_fused(
    const unsigned short* __restrict__ abuf,  // [N][256] bf16 row-major (A)
    const unsigned short* __restrict__ w1p,   // packed B layout
    const float* __restrict__ b1,
    const unsigned short* __restrict__ w2p,
    const float* __restrict__ b2,
    float* __restrict__ zout,                 // [N][256] f32 (reuses h buffer)
    float* __restrict__ gstat) {              // [512] = {sum[256], sumsq[256]}
  __shared__ __attribute__((aligned(16))) unsigned short alds[64 * HID];  // 32KB
  __shared__ __attribute__((aligned(16))) unsigned short tlds[64 * HID];  // 32KB
  int tid  = threadIdx.x;
  int wave = tid >> 5, lane = tid & 31;
  int rt    = wave >> 1;          // row tile 0..3
  int cbase = (wave & 1) * 8;     // first col tile (0 or 8)
  int mrow  = lane & 15;          // A row within tile
  int koff  = (lane >> 4) * 8;    // A K-offset for upper half-wave
  int ncol  = lane & 15;          // B/C column within tile
  int khalf = lane >> 4;          // B K-half
  long row0 = (long)blockIdx.x * 64;

  // -------- Phase 0: stage A tile into LDS (async if available) --------
  {
    const unsigned short* gsrc = abuf + row0 * HID;   // 16384 bf16 = 2048 x 16B
#if __has_builtin(__builtin_amdgcn_global_load_async_to_lds_b128) && \
    __has_builtin(__builtin_amdgcn_s_wait_asynccnt)
#pragma unroll
    for (int i = 0; i < 8; i++) {
      int chunk = tid + i * 256;                      // 16-byte units
      __builtin_amdgcn_global_load_async_to_lds_b128(
          (__attribute__((address_space(1))) b128_t*)(unsigned long long)(gsrc + chunk * 8),
          (__attribute__((address_space(3))) b128_t*)(unsigned)(unsigned long long)(alds + chunk * 8),
          0, 0);
    }
    __builtin_amdgcn_s_wait_asynccnt(0);
#else
#pragma unroll
    for (int i = 0; i < 8; i++) {
      int chunk = tid + i * 256;
      *(u16x8*)(alds + chunk * 8) = *(const u16x8*)(gsrc + chunk * 8);
    }
#endif
  }
  __syncthreads();

  v8f acc[8];

  // -------- Phase 1: GEMM1: t = relu(A @ W1 + b1) --------
#pragma unroll
  for (int i = 0; i < 8; i++) {
    float bv = b1[(cbase + i) * 16 + ncol];
#pragma unroll
    for (int j = 0; j < 8; j++) acc[i][j] = bv;
  }
#pragma unroll
  for (int kt = 0; kt < 8; kt++) {
    int kk = kt * 32;
    BfPack A;
    const u16x8* ap = (const u16x8*)(alds + ((rt * 16 + mrow) * HID + kk + koff));
    A.h[0] = ap[0];           // K = kk+koff .. +7
    A.h[1] = ap[2];           // K = kk+16+koff .. +7
#pragma unroll
    for (int i = 0; i < 8; i++) {
      int n = (cbase + i) * 16 + ncol;
      BfPack B;
      const u16x8* bp = (const u16x8*)(w1p + ((long)(kt * HID + n) * 32 + khalf * 8));
      B.h[0] = bp[0]; B.h[1] = bp[2];
      acc[i] = __builtin_amdgcn_wmma_f32_16x16x32_bf16(
          false, A.v, false, B.v, (short)0, acc[i], false, false);
    }
  }
  // relu + bf16 -> LDS t tile (C layout: vgpr j + half-wave -> row; lane%16 -> col)
#pragma unroll
  for (int i = 0; i < 8; i++) {
    int c = (cbase + i) * 16 + ncol;
#pragma unroll
    for (int j = 0; j < 8; j++) {
      int r = rt * 16 + j + (khalf << 3);
      tlds[r * HID + c] = f2bf(fmaxf(acc[i][j], 0.0f));
    }
  }
  __syncthreads();                       // t visible; alds now dead

  // BN scratch aliases the dead A-stage region.
  float* ssum = (float*)(void*)alds;     // [256]
  float* ssq  = ssum + HID;              // [256]
  ssum[tid] = 0.0f; ssq[tid] = 0.0f;
  __syncthreads();

  // -------- Phase 2: GEMM2: z2 = t @ W2 + b2 --------
#pragma unroll
  for (int i = 0; i < 8; i++) {
    float bv = b2[(cbase + i) * 16 + ncol];
#pragma unroll
    for (int j = 0; j < 8; j++) acc[i][j] = bv;
  }
#pragma unroll
  for (int kt = 0; kt < 8; kt++) {
    int kk = kt * 32;
    BfPack A;
    const u16x8* ap = (const u16x8*)(tlds + ((rt * 16 + mrow) * HID + kk + koff));
    A.h[0] = ap[0];
    A.h[1] = ap[2];
#pragma unroll
    for (int i = 0; i < 8; i++) {
      int n = (cbase + i) * 16 + ncol;
      BfPack B;
      const u16x8* bp = (const u16x8*)(w2p + ((long)(kt * HID + n) * 32 + khalf * 8));
      B.h[0] = bp[0]; B.h[1] = bp[2];
      acc[i] = __builtin_amdgcn_wmma_f32_16x16x32_bf16(
          false, A.v, false, B.v, (short)0, acc[i], false, false);
    }
  }

  // -------- epilogue: store z2 + BN partial sums --------
#pragma unroll
  for (int i = 0; i < 8; i++) {
    int c = (cbase + i) * 16 + ncol;
    float ps = 0.0f, pq = 0.0f;
#pragma unroll
    for (int j = 0; j < 8; j++) {
      long r = row0 + rt * 16 + j + (khalf << 3);
      float v = acc[i][j];
      zout[r * HID + c] = v;
      ps += v; pq += v * v;
    }
    ps += __shfl_xor(ps, 16);   // fold the two half-waves (same channel)
    pq += __shfl_xor(pq, 16);
    if (lane < 16) { atomicAdd(&ssum[c], ps); atomicAdd(&ssq[c], pq); }
  }
  __syncthreads();
  gatomic_add(&gstat[tid], ssum[tid]);
  gatomic_add(&gstat[HID + tid], ssq[tid]);
}

// ------------------------------- BN + relu ----------------------------------
__global__ void bn_apply(float* __restrict__ h, const float* __restrict__ gstat,
                         const float* __restrict__ gamma, const float* __restrict__ beta,
                         long n) {
  long i = (long)blockIdx.x * blockDim.x + threadIdx.x;
  long stride = (long)gridDim.x * blockDim.x;
  const float inv_n = 1.0f / (float)N_NODES;
  for (; i < n; i += stride) {
    int c = (int)(i & 255);
    float mu  = gstat[c] * inv_n;
    float var = gstat[HID + c] * inv_n - mu * mu;
    float sc  = gamma[c] * rsqrtf(var + BN_EPS);
    float v   = (h[i] - mu) * sc + beta[c];
    h[i] = fmaxf(v, 0.0f);
  }
}

// ------------------------------- pooling ------------------------------------
__global__ __launch_bounds__(256) void pool_sum(
    const float* __restrict__ h, const long long* __restrict__ batch,
    float* __restrict__ out, float* __restrict__ cnt) {
  int wave = threadIdx.x >> 5, lane = threadIdx.x & 31;
  long n = (long)blockIdx.x * 8 + wave;
  int g = (int)batch[n];
#pragma unroll
  for (int i = 0; i < 8; i++) {
    int c = i * 32 + lane;
    gatomic_add(&out[(long)g * HID + c], h[n * HID + c]);
  }
  if (lane == 0) gatomic_add(&cnt[g], 1.0f);
}

__global__ void pool_div(float* __restrict__ out, const float* __restrict__ cnt) {
  long i = (long)blockIdx.x * blockDim.x + threadIdx.x;  // G*HID threads
  int g = (int)(i >> 8);
  out[i] = out[i] / fmaxf(cnt[g], 1.0f);
}

// ------------------------------- launcher -----------------------------------
extern "C" void kernel_launch(void* const* d_in, const int* in_sizes, int n_in,
                              void* d_out, int out_size, void* d_ws, size_t ws_size,
                              hipStream_t stream) {
  const float*     x      = (const float*)d_in[0];
  const float*     ea     = (const float*)d_in[1];
  const long long* ei     = (const long long*)d_in[2];
  const long long* batch  = (const long long*)d_in[3];
  const float*     in_W   = (const float*)d_in[4];
  const float*     in_b   = (const float*)d_in[5];
  const float*     edge_W = (const float*)d_in[6];
  const float*     edge_b = (const float*)d_in[7];
  const float*     w1     = (const float*)d_in[8];
  const float*     b1     = (const float*)d_in[9];
  const float*     w2     = (const float*)d_in[10];
  const float*     b2     = (const float*)d_in[11];
  const float*     gamma  = (const float*)d_in[12];
  const float*     beta   = (const float*)d_in[13];
  float* out = (float*)d_out;

  char* ws = (char*)d_ws;
  size_t off = 0;
  auto alloc = [&](size_t bytes) -> char* {
    char* p = ws + off;
    off += (bytes + 255) & ~(size_t)255;
    return p;
  };
  float*          h     = (float*)alloc(sizeof(float) * (size_t)N_NODES * HID);
  float*          agg   = (float*)alloc(sizeof(float) * (size_t)N_NODES * HID);
  unsigned short* abuf  = (unsigned short*)alloc(2ull * N_NODES * HID);
  unsigned short* wp1   = (unsigned short*)alloc(2ull * N_LAYERS * HID * HID);
  unsigned short* wp2   = (unsigned short*)alloc(2ull * N_LAYERS * HID * HID);
  float*          gstat = (float*)alloc(sizeof(float) * 2 * HID);
  float*          cnt   = (float*)alloc(sizeof(float) * N_GRAPHS);

  const long nh = (long)N_NODES * HID;

  repack_w<<<N_LAYERS * HID * HID / 256, 256, 0, stream>>>(w1, w2, wp1, wp2);
  input_proj<<<N_NODES / 8, 256, 0, stream>>>(x, in_W, in_b, h);

  for (int l = 0; l < N_LAYERS; l++) {
    zero_f32<<<2048, 256, 0, stream>>>(agg, nh);
    zero_f32<<<1, 256, 0, stream>>>(gstat, 2 * HID);
    edge_msg<<<N_EDGES / 8, 256, 0, stream>>>(
        ea, ei, edge_W + (size_t)l * 3 * HID, edge_b + (size_t)l * HID, h, agg);
    add_convert<<<4096, 256, 0, stream>>>(
        (const float2*)h, (const float2*)agg, (unsigned int*)abuf, nh / 2);
    mlp_fused<<<N_NODES / 64, 256, 0, stream>>>(
        abuf, wp1 + (size_t)l * HID * HID, b1 + (size_t)l * HID,
        wp2 + (size_t)l * HID * HID, b2 + (size_t)l * HID, h, gstat);
    bn_apply<<<4096, 256, 0, stream>>>(
        h, gstat, gamma + (size_t)l * HID, beta + (size_t)l * HID, nh);
  }

  zero_f32<<<2048, 256, 0, stream>>>(out, (long)N_GRAPHS * HID);
  zero_f32<<<32, 256, 0, stream>>>(cnt, N_GRAPHS);
  pool_sum<<<N_NODES / 8, 256, 0, stream>>>(h, batch, out, cnt);
  pool_div<<<N_GRAPHS * HID / 256, 256, 0, stream>>>(out, cnt);
}
